// VectorQuantize_26113401160122
// MI455X (gfx1250) — matrix operand order repaired and checked by reading, not compiled, not run on hardware
//
#include <hip/hip_runtime.h>
#include <hip/hip_bf16.h>

typedef __bf16 bf16;
typedef __attribute__((ext_vector_type(16))) __bf16 v16bf;
typedef __attribute__((ext_vector_type(8)))  __bf16 v8bf;
typedef __attribute__((ext_vector_type(8)))  float   v8f;

namespace vqk {

constexpr int BB  = 4;
constexpr int CIN = 1024;
constexpr int TT  = 2048;
constexpr int KK  = 8192;
constexpr int DD  = 256;
constexpr int NN  = BB * TT;           // 8192
constexpr float EPSL  = 1e-10f;
constexpr float DECAY = 0.99f;

// ---- workspace layout (bytes; every size is a multiple of 256) ----
constexpr size_t OFF_ZROWS  = 0;                                    // N*CIN bf16
constexpr size_t OFF_WI     = OFF_ZROWS  + (size_t)NN * CIN * 2;    // D*CIN bf16
constexpr size_t OFF_WO     = OFF_WI     + (size_t)DD * CIN * 2;    // CIN*D bf16
constexpr size_t OFF_EB     = OFF_WO     + (size_t)CIN * DD * 2;    // K*D bf16
constexpr size_t OFF_ET     = OFF_EB     + (size_t)KK * DD * 2;     // D*K bf16
constexpr size_t OFF_ENORM  = OFF_ET     + (size_t)DD * KK * 2;     // K f32
constexpr size_t OFF_ZD     = OFF_ENORM  + (size_t)KK * 4;          // N*D f32
constexpr size_t OFF_ZDB    = OFF_ZD     + (size_t)NN * DD * 4;     // N*D bf16
constexpr size_t OFF_RNORM  = OFF_ZDB    + (size_t)NN * DD * 2;     // N f32
constexpr size_t OFF_RMAX   = OFF_RNORM  + (size_t)NN * 4;          // N f32
constexpr size_t OFF_LOGITS = OFF_RMAX   + (size_t)NN * 4;          // N*K f32
constexpr size_t OFF_Y      = OFF_LOGITS + (size_t)NN * KK * 4;     // N*K bf16
constexpr size_t OFF_COUNTS = OFF_Y      + (size_t)NN * KK * 2;     // K f32
constexpr size_t OFF_AVG    = OFF_COUNTS + (size_t)KK * 4;          // K f32
constexpr size_t OFF_ZQ     = OFF_AVG    + (size_t)KK * 4;          // N*D f32
constexpr size_t OFF_ZQB    = OFF_ZQ     + (size_t)NN * DD * 4;     // N*D bf16
constexpr size_t OFF_VQACC  = OFF_ZQB    + (size_t)NN * DD * 2;     // 1 f32 (pad 256)

// d_out layout: out(8.4M), vq, kl, perp, active, new_cluster_size(K)
constexpr size_t OUT_VQ   = (size_t)BB * CIN * TT;
constexpr size_t OUT_KL   = OUT_VQ + 1;
constexpr size_t OUT_PERP = OUT_VQ + 2;
constexpr size_t OUT_ACT  = OUT_VQ + 3;
constexpr size_t OUT_CS   = OUT_VQ + 4;

#define WMMA_BF16(a, b, c) \
  __builtin_amdgcn_wmma_f32_16x16x32_bf16(false, (a), false, (b), (short)0, (c), false, false)

// A fragment: 16x32 (MxK) bf16, row-major source with leading dim ld.
// lane<16: M=lane, K = k0+{0..15} in a[0..15]
// lane>=16: M=lane-16, K = k0+16+{0..15}
static __device__ __forceinline__
v16bf load_a_frag(const bf16* base, int ld, int row0, int k0, int lane) {
  const int row  = row0 + (lane & 15);
  const int half = lane >> 4;
  const bf16* p = base + (size_t)row * ld + k0 + half * 16;
  v8bf lo = *(const v8bf*)p;
  v8bf hi = *(const v8bf*)(p + 8);
  v16bf a;
#pragma unroll
  for (int i = 0; i < 8; ++i) { a[i] = lo[i]; a[8 + i] = hi[i]; }
  return a;
}

// B fragment: 32x16 (KxN) bf16. Source "colsrc" stores column n's K-run
// contiguously: element(k, n) = colsrc[n*ld + k].
// lane<16: N=lane, K=k0..k0+15 ; lane>=16: N=lane-16, K=k0+16..k0+31
static __device__ __forceinline__
v16bf load_b_frag(const bf16* colsrc, int ld, int n0, int k0, int lane) {
  const int col  = n0 + (lane & 15);
  const int half = lane >> 4;
  return *(const v16bf*)(colsrc + (size_t)col * ld + k0 + half * 16);
}

static __device__ __forceinline__ void atomicMaxFloat(float* addr, float v) {
  if (v >= 0.0f) atomicMax((int*)addr, __float_as_int(v));
  else           atomicMin((unsigned int*)addr, __float_as_uint(v));
}

// ---------------- prep kernels ----------------

__global__ void k_init(float* rmax, float* counts, float* avg, float* vqacc) {
  int idx = blockIdx.x * 256 + threadIdx.x;
  if (idx < NN) rmax[idx] = -__builtin_inff();
  if (idx < KK) { counts[idx] = 0.0f; avg[idx] = 0.0f; }
  if (idx == 0) vqacc[0] = 0.0f;
}

__global__ void k_cvtw(const float* Wi, const float* Wo, bf16* wib, bf16* wob) {
  int idx = blockIdx.x * 256 + threadIdx.x;     // D*CIN == CIN*D elements
  wib[idx] = (bf16)Wi[idx];
  wob[idx] = (bf16)Wo[idx];
}

// z (B, CIN, T) f32 -> zrows (N=B*T, CIN) bf16   (LDS-tiled transpose)
__global__ void k_tz(const float* z, bf16* zrows) {
  __shared__ float tile[32][33];
  const int t0 = blockIdx.x * 32;
  const int i0 = blockIdx.y * 32;
  const int b  = blockIdx.z;
  const int tx = threadIdx.x;   // 0..31
  const int ty = threadIdx.y;   // 0..7
#pragma unroll
  for (int j = 0; j < 4; ++j) {
    int i = i0 + ty + j * 8;
    tile[ty + j * 8][tx] = z[(size_t)b * CIN * TT + (size_t)i * TT + t0 + tx];
  }
  __syncthreads();
#pragma unroll
  for (int j = 0; j < 4; ++j) {
    int t = t0 + ty + j * 8;
    zrows[((size_t)(b * TT + t)) * CIN + i0 + tx] = (bf16)tile[tx][ty + j * 8];
  }
}

// embedding (K, D) f32 -> e_bf16 (K,D) and eT_bf16 (D,K)
__global__ void k_pe(const float* e, bf16* eb, bf16* eT) {
  __shared__ float tile[32][33];
  const int d0 = blockIdx.x * 32;
  const int k0 = blockIdx.y * 32;
  const int tx = threadIdx.x, ty = threadIdx.y;
#pragma unroll
  for (int j = 0; j < 4; ++j) {
    int k = k0 + ty + j * 8;
    float v = e[(size_t)k * DD + d0 + tx];
    eb[(size_t)k * DD + d0 + tx] = (bf16)v;
    tile[ty + j * 8][tx] = v;
  }
  __syncthreads();
#pragma unroll
  for (int j = 0; j < 4; ++j) {
    int d = d0 + ty + j * 8;
    eT[(size_t)d * KK + k0 + tx] = (bf16)tile[tx][ty + j * 8];
  }
}

// squared L2 norm of each 256-wide row; one wave per row
__global__ void k_sqnorm(const float* src, float* dst) {
  const int lane = threadIdx.x & 31;
  const int wave = threadIdx.x >> 5;
  const int row  = blockIdx.x * 8 + wave;
  float s = 0.0f;
#pragma unroll
  for (int j = 0; j < 8; ++j) {
    float v = src[(size_t)row * 256 + lane + j * 32];
    s += v * v;
  }
#pragma unroll
  for (int off = 16; off >= 1; off >>= 1) s += __shfl_xor(s, off, 32);
  if (lane == 0) dst[row] = s;
}

// ---------------- WMMA GEMM kernels ----------------

// z_d = zrows(N x CIN) * Wi^T + bi   -> zd f32 + bf16, full D per block
__global__ void k_inproj(const bf16* zrows, const bf16* wib, const float* bi,
                         float* zd, bf16* zdb) {
  const int lane = threadIdx.x & 31;
  const int wave = threadIdx.x >> 5;
  const int rowBase = blockIdx.x * 16;
  const int n0a = wave * 32, n0b = n0a + 16;
  v8f acc0 = {}, acc1 = {};
  for (int ii = 0; ii < CIN; ii += 32) {
    v16bf a  = load_a_frag(zrows, CIN, rowBase, ii, lane);
    v16bf b0 = load_b_frag(wib,   CIN, n0a,     ii, lane);
    v16bf b1 = load_b_frag(wib,   CIN, n0b,     ii, lane);
    acc0 = WMMA_BF16(a, b0, acc0);
    acc1 = WMMA_BF16(a, b1, acc1);
  }
  const int half = lane >> 4;
  const int d0 = n0a + (lane & 15), d1 = n0b + (lane & 15);
  const float bi0 = bi[d0], bi1 = bi[d1];
#pragma unroll
  for (int r = 0; r < 8; ++r) {
    int row = rowBase + r + 8 * half;
    float v0 = acc0[r] + bi0;
    float v1 = acc1[r] + bi1;
    size_t i0 = (size_t)row * DD + d0, i1 = (size_t)row * DD + d1;
    zd[i0] = v0; zdb[i0] = (bf16)v0;
    zd[i1] = v1; zdb[i1] = (bf16)v1;
  }
}

// logits = -(||z||^2 - 2 z.e + ||e||^2) + gumbel(u) ; per-row max via atomics
__global__ void k_dist(const bf16* zdb, const bf16* eb, const float* rnorm,
                       const float* enorm, const float* u,
                       float* logits, float* rmax) {
  const int lane = threadIdx.x & 31;
  const int wave = threadIdx.x >> 5;
  const int rowBase = blockIdx.y * 16;
  const int kBase   = blockIdx.x * 128 + wave * 16;
  const int half = lane >> 4;
  const int kcol = kBase + (lane & 15);
  __builtin_prefetch(&u[(size_t)(rowBase + 8 * half) * KK + kcol], 0, 1);
  v8f acc = {};
#pragma unroll
  for (int dd = 0; dd < DD; dd += 32) {
    v16bf a = load_a_frag(zdb, DD, rowBase, dd, lane);
    v16bf b = load_b_frag(eb,  DD, kBase,   dd, lane);
    acc = WMMA_BF16(a, b, acc);
  }
  const float en = enorm[kcol];
  float lv[8];
#pragma unroll
  for (int r = 0; r < 8; ++r) {
    int row = rowBase + r + 8 * half;
    float dist = rnorm[row] + en - 2.0f * acc[r];
    float uu = u[(size_t)row * KK + kcol];
    float g  = -__logf(-__logf(uu + EPSL) + EPSL);
    float l  = g - dist;
    logits[(size_t)row * KK + kcol] = l;
    lv[r] = l;
  }
#pragma unroll
  for (int r = 0; r < 8; ++r) {
    float v = lv[r];
#pragma unroll
    for (int off = 8; off >= 1; off >>= 1)
      v = fmaxf(v, __shfl_xor(v, off, 16));
    if ((lane & 15) == 0)
      atomicMaxFloat(&rmax[rowBase + r + 8 * half], v);
  }
}

// per-row softmax with known max; writes normalized probs bf16; argmax->counts
__global__ void k_softmax(const float* logits, const float* rmax,
                          bf16* y, float* counts) {
  const int row = blockIdx.x;
  const int tid = threadIdx.x;
  __shared__ float srow[KK];        // 32 KB of the 320 KB WGP LDS
  __shared__ float rsum[256];
  __shared__ float rbv[256];
  __shared__ int   rbi[256];
  const float m = rmax[row];
  float sum = 0.0f, bestv = -__builtin_inff();
  int besti = 0;
  for (int k = tid; k < KK; k += 256) {
    float l = logits[(size_t)row * KK + k];
    float e = __expf(l - m);
    srow[k] = e;
    sum += e;
    if (l > bestv) { bestv = l; besti = k; }
  }
  rsum[tid] = sum; rbv[tid] = bestv; rbi[tid] = besti;
  __syncthreads();
  for (int s = 128; s > 0; s >>= 1) {
    if (tid < s) {
      rsum[tid] += rsum[tid + s];
      float ov = rbv[tid + s]; int oi = rbi[tid + s];
      if (ov > rbv[tid] || (ov == rbv[tid] && oi < rbi[tid])) {
        rbv[tid] = ov; rbi[tid] = oi;
      }
    }
    __syncthreads();
  }
  const float zinv = 1.0f / rsum[0];
  for (int k = tid; k < KK; k += 256)
    y[(size_t)row * KK + k] = (bf16)(srow[k] * zinv);
  if (tid == 0) atomicAdd(&counts[rbi[0]], 1.0f);
}

// avg[k] += partial column sums of y (finalize divides by N)
__global__ void k_colsum(const bf16* y, float* avg) {
  const int k  = blockIdx.x * 256 + threadIdx.x;
  const int n0 = blockIdx.y * (NN / 16);
  float s = 0.0f;
  for (int n = n0; n < n0 + NN / 16; ++n) s += (float)y[(size_t)n * KK + k];
  atomicAdd(&avg[k], s);
}

// z_q = Y(N x K) * E(K x D) via eT; fused vq-loss partial
__global__ void k_zq(const bf16* y, const bf16* eT, const float* zd,
                     float* zq, bf16* zqb, float* vqacc) {
  const int lane = threadIdx.x & 31;
  const int wave = threadIdx.x >> 5;
  const int rowBase = blockIdx.x * 16;
  const int n0a = wave * 32, n0b = n0a + 16;
  v8f acc0 = {}, acc1 = {};
  for (int kk = 0; kk < KK; kk += 32) {
    v16bf a  = load_a_frag(y,  KK, rowBase, kk, lane);
    v16bf b0 = load_b_frag(eT, KK, n0a,     kk, lane);
    v16bf b1 = load_b_frag(eT, KK, n0b,     kk, lane);
    acc0 = WMMA_BF16(a, b0, acc0);
    acc1 = WMMA_BF16(a, b1, acc1);
  }
  const int half = lane >> 4;
  const int d0 = n0a + (lane & 15), d1 = n0b + (lane & 15);
  float local = 0.0f;
#pragma unroll
  for (int r = 0; r < 8; ++r) {
    int row = rowBase + r + 8 * half;
    size_t i0 = (size_t)row * DD + d0, i1 = (size_t)row * DD + d1;
    float v0 = acc0[r], v1 = acc1[r];
    zq[i0] = v0; zqb[i0] = (bf16)v0;
    zq[i1] = v1; zqb[i1] = (bf16)v1;
    float a0 = zd[i0] - v0, a1 = zd[i1] - v1;
    local += a0 * a0 + a1 * a1;
  }
  __shared__ float red[256];
  red[threadIdx.x] = local;
  __syncthreads();
  for (int s = 128; s > 0; s >>= 1) {
    if (threadIdx.x < s) red[threadIdx.x] += red[threadIdx.x + s];
    __syncthreads();
  }
  if (threadIdx.x == 0) atomicAdd(vqacc, red[0]);
}

// out(b,o,t) = z_q(N x D) * Wo^T + bo
__global__ void k_out(const bf16* zqb, const bf16* wob, const float* bo,
                      float* out) {
  const int lane = threadIdx.x & 31;
  const int wave = threadIdx.x >> 5;
  const int rowBase = blockIdx.y * 16;
  const int oBase   = blockIdx.x * 128 + wave * 16;
  v8f acc = {};
#pragma unroll
  for (int dd = 0; dd < DD; dd += 32) {
    v16bf a = load_a_frag(zqb, DD, rowBase, dd, lane);
    v16bf b = load_b_frag(wob, DD, oBase,   dd, lane);
    acc = WMMA_BF16(a, b, acc);
  }
  const int half = lane >> 4;
  const int o = oBase + (lane & 15);
  const float bov = bo[o];
#pragma unroll
  for (int r = 0; r < 8; ++r) {
    int n = rowBase + r + 8 * half;
    int b = n >> 11;            // / T (T = 2048)
    int t = n & (TT - 1);
    out[(size_t)b * CIN * TT + (size_t)o * TT + t] = acc[r] + bov;
  }
}

// KL, perplexity, active_num, EMA cluster size, scalar writes
__global__ void k_finalize(const float* avg, const float* counts,
                           const float* cs, const float* vqacc, float* out) {
  const int tid = threadIdx.x;
  __shared__ float rA[256], rB[256], rC[256];
  float skl = 0.0f, sh = 0.0f, sact = 0.0f;
  const float invN = 1.0f / (float)NN;
  for (int k = tid; k < KK; k += 256) {
    float ap = avg[k] * invN;
    skl += __logf(ap + EPSL);
    float p = counts[k] * invN;
    sh  += p * __logf(p + EPSL);
    sact += (cs[k] > 2.0f) ? 1.0f : 0.0f;
    out[OUT_CS + k] = cs[k] * DECAY + counts[k] * (1.0f - DECAY);
  }
  rA[tid] = skl; rB[tid] = sh; rC[tid] = sact;
  __syncthreads();
  for (int s = 128; s > 0; s >>= 1) {
    if (tid < s) { rA[tid] += rA[tid + s]; rB[tid] += rB[tid + s]; rC[tid] += rC[tid + s]; }
    __syncthreads();
  }
  if (tid == 0) {
    const float invK = 1.0f / (float)KK;
    const float logInvK = __logf(invK);
    out[OUT_VQ]   = vqacc[0] / (float)((size_t)NN * DD);
    out[OUT_KL]   = (float)NN * invK * ((float)KK * logInvK - rA[0]);
    out[OUT_PERP] = __expf(-rB[0]);
    out[OUT_ACT]  = rC[0];
  }
}

} // namespace vqk

extern "C" void kernel_launch(void* const* d_in, const int* in_sizes, int n_in,
                              void* d_out, int out_size, void* d_ws, size_t ws_size,
                              hipStream_t stream) {
  using namespace vqk;
  (void)in_sizes; (void)n_in; (void)out_size; (void)ws_size;

  const float* z   = (const float*)d_in[0];
  const float* u   = (const float*)d_in[1];
  const float* Wi  = (const float*)d_in[2];
  const float* bi  = (const float*)d_in[3];
  const float* Wo  = (const float*)d_in[4];
  const float* bo  = (const float*)d_in[5];
  const float* emb = (const float*)d_in[6];
  const float* cs  = (const float*)d_in[7];
  float* out = (float*)d_out;
  char*  ws  = (char*)d_ws;

  bf16*  zrows  = (bf16*) (ws + OFF_ZROWS);
  bf16*  wib    = (bf16*) (ws + OFF_WI);
  bf16*  wob    = (bf16*) (ws + OFF_WO);
  bf16*  eb     = (bf16*) (ws + OFF_EB);
  bf16*  eT     = (bf16*) (ws + OFF_ET);
  float* enorm  = (float*)(ws + OFF_ENORM);
  float* zd     = (float*)(ws + OFF_ZD);
  bf16*  zdb    = (bf16*) (ws + OFF_ZDB);
  float* rnorm  = (float*)(ws + OFF_RNORM);
  float* rmax   = (float*)(ws + OFF_RMAX);
  float* logits = (float*)(ws + OFF_LOGITS);
  bf16*  y      = (bf16*) (ws + OFF_Y);
  float* counts = (float*)(ws + OFF_COUNTS);
  float* avg    = (float*)(ws + OFF_AVG);
  float* zq     = (float*)(ws + OFF_ZQ);
  bf16*  zqb    = (bf16*) (ws + OFF_ZQB);
  float* vqacc  = (float*)(ws + OFF_VQACC);

  k_init   <<<KK / 256, 256, 0, stream>>>(rmax, counts, avg, vqacc);
  k_cvtw   <<<(DD * CIN) / 256, 256, 0, stream>>>(Wi, Wo, wib, wob);
  k_tz     <<<dim3(TT / 32, CIN / 32, BB), dim3(32, 8), 0, stream>>>(z, zrows);
  k_pe     <<<dim3(DD / 32, KK / 32), dim3(32, 8), 0, stream>>>(emb, eb, eT);
  k_sqnorm <<<KK / 8, 256, 0, stream>>>(emb, enorm);

  k_inproj <<<NN / 16, 256, 0, stream>>>(zrows, wib, bi, zd, zdb);
  k_sqnorm <<<NN / 8, 256, 0, stream>>>(zd, rnorm);

  k_dist   <<<dim3(KK / 128, NN / 16), 256, 0, stream>>>(zdb, eb, rnorm, enorm,
                                                         u, logits, rmax);
  k_softmax<<<NN, 256, 0, stream>>>(logits, rmax, y, counts);
  k_colsum <<<dim3(KK / 256, 16), 256, 0, stream>>>(y, avg);

  k_zq     <<<NN / 16, 256, 0, stream>>>(y, eT, zd, zq, zqb, vqacc);
  k_out    <<<dim3(CIN / 128, NN / 16), 256, 0, stream>>>(zqb, wob, bo, out);
  k_finalize<<<1, 256, 0, stream>>>(avg, counts, cs, vqacc, out);
}
